// HybridAttentionLSTMAutoencoder_12927851561255
// MI455X (gfx1250) — matrix operand order repaired and checked by reading, not compiled, not run on hardware
//
#include <hip/hip_runtime.h>
#include <hip/hip_bf16.h>
#include <math.h>

// ---------------------------------------------------------------------------
// HybridAttentionLSTMAutoencoder on MI455X (gfx1250, wave32, WMMA).
//
// bf16 activations/weights end-to-end (fp32 accumulation in WMMA, fp32
// epilogues); tile staging uses the CDNA5 async paths:
// GLOBAL_LOAD_ASYNC_TO_LDS_B128 (ASYNCcnt) for GEMM/attention tiles and the
// Tensor Data Mover (TENSORcnt) for the LDS-resident LSTM Whh. Both are
// __has_builtin-guarded with a synchronous vector-copy fallback.
// ---------------------------------------------------------------------------

#define Bsz 256
#define Ssz 256
#define Fsz 64
#define Hsz 128
#define Esz 256
#define NHsz 4
#define Mtok (Bsz * Ssz)          // 65536 tokens

typedef __attribute__((ext_vector_type(16))) __bf16 v16bf;
typedef __attribute__((ext_vector_type(8)))  __bf16 v8bf;
typedef __attribute__((ext_vector_type(8)))  float  v8f;
typedef __attribute__((ext_vector_type(4)))  unsigned int u32x4;
typedef __attribute__((ext_vector_type(8)))  int          i32x8;
typedef __attribute__((ext_vector_type(4)))  int          i32x4;
typedef int v4i_gcc __attribute__((vector_size(16)));   // builtin's pointee type
typedef __attribute__((address_space(1))) v4i_gcc* gptr_v4i;
typedef __attribute__((address_space(3))) v4i_gcc* lptr_v4i;

#if defined(__has_builtin)
#  if __has_builtin(__builtin_amdgcn_global_load_async_to_lds_b128) && \
      __has_builtin(__builtin_amdgcn_s_wait_asynccnt)
#    define HAVE_ASYNC_LDS 1
#  else
#    define HAVE_ASYNC_LDS 0
#  endif
#  if __has_builtin(__builtin_amdgcn_tensor_load_to_lds) && \
      __has_builtin(__builtin_amdgcn_s_wait_tensorcnt)
#    define HAVE_TDM 1
#  else
#    define HAVE_TDM 0
#  endif
#else
#  define HAVE_ASYNC_LDS 0
#  define HAVE_TDM 0
#endif

__device__ __forceinline__ __bf16 f2bf(float f) {
    union { float f; unsigned u; } x; x.f = f;
    unsigned r = (x.u + 0x7FFFu + ((x.u >> 16) & 1u)) >> 16;   // RNE
    union { unsigned short s; __bf16 b; } y; y.s = (unsigned short)r;
    return y.b;
}
__device__ __forceinline__ float bf2f(__bf16 b) {
    union { unsigned short s; __bf16 b; } y; y.b = b;
    union { unsigned u; float f; } x; x.u = ((unsigned)y.s) << 16;
    return x.f;
}
__device__ __forceinline__ float sigmoidf_(float x) { return 1.0f / (1.0f + __expf(-x)); }

__device__ __forceinline__ v8f zero8() {
    v8f z;
#pragma unroll
    for (int i = 0; i < 8; ++i) z[i] = 0.0f;
    return z;
}

// 16-byte global->LDS copy: async (gfx1250 GLOBAL_LOAD_ASYNC_TO_LDS_B128,
// signature (V4i addrspace(1)*, V4i addrspace(3)*, Imm offset, Imm cpol)) or
// synchronous vector copy fallback.
__device__ __forceinline__ void cp16(const __bf16* g, __bf16* l) {
#if HAVE_ASYNC_LDS
    __builtin_amdgcn_global_load_async_to_lds_b128((gptr_v4i)g, (lptr_v4i)l, 0, 0);
#else
    *(v8bf*)l = *(const v8bf*)g;
#endif
}
__device__ __forceinline__ void cp_wait() {
#if HAVE_ASYNC_LDS
    __builtin_amdgcn_s_wait_asynccnt(0);
#endif
}

// A-fragment (16Mx32K bf16): lanes<16 hold K {0..7,16..23}, lanes>=16 K {8..15,24..31}.
__device__ __forceinline__ v16bf load_a_frag(const __bf16* row, int lane) {
    const int kh = ((lane >> 4) << 3);
    const v8bf lo = *(const v8bf*)(row + kh);
    const v8bf hi = *(const v8bf*)(row + 16 + kh);
    v16bf a;
#pragma unroll
    for (int j = 0; j < 8; ++j) { a[j] = lo[j]; a[j + 8] = hi[j]; }
    return a;
}
// B-fragment (32Kx16N bf16): lane n=L%16, lanes<16 hold K {0..7,16..23}, lanes>=16 K {8..15,24..31}.
// `nrow` points at element [n][0] of B^T ([N][K]) data in a 32-wide K window.
__device__ __forceinline__ v16bf load_b_frag(const __bf16* nrow, int lane) {
    const int kh = ((lane >> 4) << 3);
    const v8bf lo = *(const v8bf*)(nrow + kh);
    const v8bf hi = *(const v8bf*)(nrow + 16 + kh);
    v16bf b;
#pragma unroll
    for (int j = 0; j < 8; ++j) { b[j] = lo[j]; b[j + 8] = hi[j]; }
    return b;
}
__device__ __forceinline__ v8f wmma_bf16(v16bf a, v16bf b, v8f c) {
    return __builtin_amdgcn_wmma_f32_16x16x32_bf16(false, a, false, b, (short)0, c, false, false);
}

// ---------------------------------------------------------------------------
// bf16 GEMM:  C[M,N] = act( A[M,K] @ W[N,K]^T + b1 + b2 ), fp32 accum.
// Requires M%64==0, N%64==0, K%32==0.  Block: 128 threads (4 waves),
// 64x64 tile, 32x32 per wave (2x2 WMMA).  C written bf16 or fp32.
// ---------------------------------------------------------------------------
__global__ void __launch_bounds__(128)
gemm_wmma(const __bf16* __restrict__ A, int lda,
          const __bf16* __restrict__ W, // [N,K] bf16 (torch Linear layout)
          const float* __restrict__ b1, const float* __restrict__ b2,
          void* __restrict__ Cv, int ldc, int coff,
          int Ksz_, int relu, int out_bf16) {
    __shared__ __align__(64) __bf16 As[64][32];
    __shared__ __align__(64) __bf16 Ws[64][32];

    const int tid  = threadIdx.x;
    const int lane = tid & 31;
    const int w    = tid >> 5;
    const int wm   = (w >> 1) * 32;
    const int wn   = (w & 1) * 32;
    const int mb   = blockIdx.y * 64;
    const int nb   = blockIdx.x * 64;

    v8f acc[2][2];
#pragma unroll
    for (int mi = 0; mi < 2; ++mi)
#pragma unroll
        for (int ni = 0; ni < 2; ++ni) acc[mi][ni] = zero8();

    for (int k0 = 0; k0 < Ksz_; k0 += 32) {
        // stage 64x32 bf16 tiles: 256 16B chunks per tile, 2 per thread
#pragma unroll
        for (int rep = 0; rep < 2; ++rep) {
            const int c = tid + rep * 128;
            const int r = c >> 2, seg = (c & 3) * 8;
            cp16(A + (size_t)(mb + r) * lda + k0 + seg, &As[r][seg]);
            cp16(W + (size_t)(nb + r) * Ksz_ + k0 + seg, &Ws[r][seg]);
        }
        cp_wait();
        __syncthreads();

        v16bf af[2], bf[2];
#pragma unroll
        for (int mi = 0; mi < 2; ++mi)
            af[mi] = load_a_frag(&As[wm + mi * 16 + (lane & 15)][0], lane);
#pragma unroll
        for (int ni = 0; ni < 2; ++ni)
            bf[ni] = load_b_frag(&Ws[wn + ni * 16 + (lane & 15)][0], lane);
#pragma unroll
        for (int mi = 0; mi < 2; ++mi)
#pragma unroll
            for (int ni = 0; ni < 2; ++ni)
                acc[mi][ni] = wmma_bf16(af[mi], bf[ni], acc[mi][ni]);
        __syncthreads();
    }

#pragma unroll
    for (int mi = 0; mi < 2; ++mi)
#pragma unroll
        for (int ni = 0; ni < 2; ++ni) {
            const int col = nb + wn + ni * 16 + (lane & 15);
            float bias = 0.0f;
            if (b1) bias += b1[col];
            if (b2) bias += b2[col];
#pragma unroll
            for (int r = 0; r < 8; ++r) {
                const int row = mb + wm + mi * 16 + ((lane >> 4) << 3) + r;
                float v = acc[mi][ni][r] + bias;
                if (relu) v = fmaxf(v, 0.0f);
                const size_t idx = (size_t)row * ldc + coff + col;
                if (out_bf16) ((__bf16*)Cv)[idx] = f2bf(v);
                else          ((float*)Cv)[idx]  = v;
            }
        }
}

// ---------------------------------------------------------------------------
// LSTM scan. gates[B,S,512] (bf16) = x@Wih^T + bih + bhh. One workgroup per 16
// batch rows; Whh [512,128] bf16 resident in LDS (128KB), loaded via the
// Tensor Data Mover when available. h (bf16) / c (f32) stay on-chip. Output h
// written bf16 at column offset coff, row stride ldo.
// Dynamic LDS: Whh 131072 | Hs 4096 | Gs 32768 | Cs 8192 = 176128 bytes.
// ---------------------------------------------------------------------------
__global__ void __launch_bounds__(256)
lstm_scan(const __bf16* __restrict__ gates, const __bf16* __restrict__ Whh,
          __bf16* __restrict__ out, int ldo, int coff, int reverse) {
    extern __shared__ __align__(64) char smem[];
    __bf16* WhhS = (__bf16*)smem;                          // [512][128]
    __bf16* Hs   = (__bf16*)(smem + 131072);               // [16][128]
    float*  Gs   = (float*)(smem + 131072 + 4096);         // [16][512]
    float*  Cs   = (float*)(smem + 131072 + 4096 + 32768); // [16][128]

    const int tid  = threadIdx.x;
    const int lane = tid & 31;
    const int w    = tid >> 5;          // 0..7 -> N slice [w*64, w*64+64)
    const int b0   = blockIdx.x * 16;

#if HAVE_TDM
    if (tid < 32) {
        // 2-D TDM descriptor: tile 128 x 512 of 2-byte elements -> LDS base.
        const size_t ga = (size_t)Whh;
        u32x4 g0;
        g0[0] = 1u;                                        // count=1, user mode
        g0[1] = __builtin_amdgcn_groupstaticsize();        // lds_addr (dyn base)
        g0[2] = (unsigned)(ga & 0xFFFFFFFFu);              // global_addr[31:0]
        g0[3] = (unsigned)((ga >> 32) & 0x01FFFFFFu) | (2u << 30); // [56:32]|type=2
        i32x8 g1 = { 0x00010000,          // wg_mask=0, data_size=1 (2B)
                     (int)(128u << 16),   // tensor_dim0 = 128 (low 16 in [31:16])
                     (int)(512u << 16),   // tensor_dim1 = 512
                     (int)(128u << 16),   // tile_dim0 = 128 ([127:112])
                     512,                 // tile_dim1 = 512 ([143:128])
                     128,                 // tensor_dim0_stride = 128
                     0, 0 };
        i32x4 gz = { 0, 0, 0, 0 };
#if __clang_major__ >= 23
        i32x8 gz8 = { 0, 0, 0, 0, 0, 0, 0, 0 };
        __builtin_amdgcn_tensor_load_to_lds(g0, g1, gz, gz, gz8, 0);
#else
        __builtin_amdgcn_tensor_load_to_lds(g0, g1, gz, gz, 0);
#endif
        __builtin_amdgcn_s_wait_tensorcnt(0);
    }
#else
    for (int e = tid * 8; e < 512 * 128; e += 256 * 8)
        *(v8bf*)&WhhS[e] = *(const v8bf*)(Whh + e);
#endif
    for (int e = tid; e < 16 * 128; e += 256) { Hs[e] = f2bf(0.0f); Cs[e] = 0.0f; }
    __syncthreads();

    for (int it = 0; it < Ssz; ++it) {
        const int t = reverse ? (Ssz - 1 - it) : it;

        // ---- G = h @ Whh^T for this wave's 64 output columns ----
        v8f acc[4];
#pragma unroll
        for (int nt = 0; nt < 4; ++nt) acc[nt] = zero8();
#pragma unroll
        for (int ks = 0; ks < 4; ++ks) {
            const v16bf af = load_a_frag(&Hs[(lane & 15) * 128 + ks * 32], lane);
#pragma unroll
            for (int nt = 0; nt < 4; ++nt) {
                const int n = w * 64 + nt * 16 + (lane & 15);
                const v16bf bf = load_b_frag(&WhhS[n * 128 + ks * 32], lane);
                acc[nt] = wmma_bf16(af, bf, acc[nt]);
            }
        }
#pragma unroll
        for (int nt = 0; nt < 4; ++nt) {
            const int n = w * 64 + nt * 16 + (lane & 15);
#pragma unroll
            for (int r = 0; r < 8; ++r) {
                const int m = ((lane >> 4) << 3) + r;
                const size_t gi = ((size_t)(b0 + m) * Ssz + t) * 512 + n;
                Gs[m * 512 + n] = acc[nt][r] + bf2f(gates[gi]);
            }
        }
        __syncthreads();

        // ---- gate nonlinearity: 16 rows x 128 cols, 8 per thread ----
#pragma unroll
        for (int q = 0; q < 8; ++q) {
            const int e = tid + q * 256;
            const int row = e >> 7, hc = e & 127;
            const float gi = Gs[row * 512 + hc];
            const float gf = Gs[row * 512 + 128 + hc];
            const float gg = Gs[row * 512 + 256 + hc];
            const float go = Gs[row * 512 + 384 + hc];
            float c = sigmoidf_(gf) * Cs[row * 128 + hc] + sigmoidf_(gi) * tanhf(gg);
            float h = sigmoidf_(go) * tanhf(c);
            Cs[row * 128 + hc] = c;
            Hs[row * 128 + hc] = f2bf(h);
            out[((size_t)(b0 + row) * Ssz + t) * ldo + coff + hc] = f2bf(h);
        }
        __syncthreads();
    }
}

// ---------------------------------------------------------------------------
// Fused attention for one (b, head, q-block of 128 rows). qkv is bf16 with the
// 1/sqrt(dh) factor pre-folded into the q projection weights. K staged via
// async-to-LDS; V transposed manually. Row softmax via wave32 shfl_xor.
// Probabilities -> LDS (bf16) + atomicAdd(0.15*p) into comb; then ao = P@V.
// Dynamic LDS: Kt 32768 | Vt 32768 | Wp 65536 = 131072 bytes.
// ---------------------------------------------------------------------------
__global__ void __launch_bounds__(256)
attn_fused(const __bf16* __restrict__ qkv, __bf16* __restrict__ ao,
           float* __restrict__ comb) {
    extern __shared__ __align__(64) char smem[];
    __bf16* Kt = (__bf16*)smem;                 // [256][64]
    __bf16* Vt = (__bf16*)(smem + 32768);       // [64][256]  (V transposed)
    __bf16* Wp = (__bf16*)(smem + 65536);       // [8 waves][16][256]

    const int tid  = threadIdx.x;
    const int lane = tid & 31;
    const int w    = tid >> 5;
    const int qblk = blockIdx.x;        // 0..1
    const int h    = blockIdx.y;        // 0..3
    const int b    = blockIdx.z;        // 0..255

    // ---- stage K (async path: 16B chunks) and V^T (transpose, manual) ----
#pragma unroll
    for (int rep = 0; rep < 8; ++rep) {
        const int c = tid + rep * 256;          // 2048 chunks
        const int s = c >> 3, seg = (c & 7) * 8;
        cp16(qkv + (size_t)(b * Ssz + s) * (3 * Esz) + Esz + h * 64 + seg,
             &Kt[s * 64 + seg]);
    }
#pragma unroll 4
    for (int i = 0; i < 64; ++i) {
        const int e = tid + i * 256;
        const int s = e >> 6, d = e & 63;
        Vt[d * 256 + s] = qkv[(size_t)(b * Ssz + s) * (3 * Esz) + 2 * Esz + h * 64 + d];
    }
    cp_wait();
    __syncthreads();

    // ---- q fragments (scale folded into w_in/b_in) ----
    const int m0 = qblk * 128 + w * 16;
    const __bf16* qrow = qkv + (size_t)(b * Ssz + m0 + (lane & 15)) * (3 * Esz) + h * 64;
    v16bf qf[2];
#pragma unroll
    for (int ks = 0; ks < 2; ++ks) {
        const int kh = ((lane >> 4) << 3);
#pragma unroll
        for (int j = 0; j < 8; ++j) {
            qf[ks][j]     = qrow[ks * 32 + kh + j];
            qf[ks][j + 8] = qrow[ks * 32 + 16 + kh + j];
        }
    }

    // ---- scores ----
    v8f sc[16];
#pragma unroll
    for (int nt = 0; nt < 16; ++nt) {
        sc[nt] = zero8();
#pragma unroll
        for (int ks = 0; ks < 2; ++ks) {
            const v16bf kf = load_b_frag(&Kt[(nt * 16 + (lane & 15)) * 64 + ks * 32], lane);
            sc[nt] = wmma_bf16(qf[ks], kf, sc[nt]);
        }
    }

    // ---- row softmax ----
#pragma unroll
    for (int r = 0; r < 8; ++r) {
        float mx = -3.0e38f;
#pragma unroll
        for (int nt = 0; nt < 16; ++nt) mx = fmaxf(mx, sc[nt][r]);
        mx = fmaxf(mx, __shfl_xor(mx, 1, 32));
        mx = fmaxf(mx, __shfl_xor(mx, 2, 32));
        mx = fmaxf(mx, __shfl_xor(mx, 4, 32));
        mx = fmaxf(mx, __shfl_xor(mx, 8, 32));
        float sum = 0.0f;
#pragma unroll
        for (int nt = 0; nt < 16; ++nt) { const float e0 = __expf(sc[nt][r] - mx); sc[nt][r] = e0; sum += e0; }
        sum += __shfl_xor(sum, 1, 32);
        sum += __shfl_xor(sum, 2, 32);
        sum += __shfl_xor(sum, 4, 32);
        sum += __shfl_xor(sum, 8, 32);
        const float inv = 1.0f / sum;
#pragma unroll
        for (int nt = 0; nt < 16; ++nt) sc[nt][r] *= inv;
    }

    // ---- probabilities -> LDS + averaged-attention accumulation ----
#pragma unroll
    for (int nt = 0; nt < 16; ++nt) {
        const int col = nt * 16 + (lane & 15);
#pragma unroll
        for (int r = 0; r < 8; ++r) {
            const int mr = ((lane >> 4) << 3) + r;
            const float p = sc[nt][r];
            Wp[(w * 16 + mr) * 256 + col] = f2bf(p);
            atomicAdd(&comb[((size_t)b * Ssz + (m0 + mr)) * Ssz + col], 0.15f * p);
        }
    }
    __syncthreads();

    // ---- ao = P @ V  (K=256, N=64) ----
    v8f oa[4];
#pragma unroll
    for (int nt = 0; nt < 4; ++nt) oa[nt] = zero8();
#pragma unroll
    for (int ks = 0; ks < 8; ++ks) {
        const v16bf pf = load_a_frag(&Wp[(w * 16 + (lane & 15)) * 256 + ks * 32], lane);
#pragma unroll
        for (int nt = 0; nt < 4; ++nt) {
            const v16bf vf = load_b_frag(&Vt[(nt * 16 + (lane & 15)) * 256 + ks * 32], lane);
            oa[nt] = wmma_bf16(pf, vf, oa[nt]);
        }
    }
#pragma unroll
    for (int nt = 0; nt < 4; ++nt) {
        const int col = h * 64 + nt * 16 + (lane & 15);
#pragma unroll
        for (int r = 0; r < 8; ++r) {
            const int row = m0 + ((lane >> 4) << 3) + r;
            ao[((size_t)b * Ssz + row) * Esz + col] = f2bf(oa[nt][r]);
        }
    }
}

// ---------------------------------------------------------------------------
__global__ void __launch_bounds__(256)
bn_stats(const __bf16* __restrict__ d, const float* __restrict__ gamma,
         const float* __restrict__ beta, float* __restrict__ st) {
    __shared__ float s1[256], s2[256];
    const int c = blockIdx.x, tid = threadIdx.x;
    float a = 0.0f, b = 0.0f;
    for (int i = tid; i < Mtok; i += 256) {
        const float v = bf2f(d[(size_t)i * Hsz + c]);
        a += v; b += v * v;
    }
    s1[tid] = a; s2[tid] = b; __syncthreads();
    for (int s = 128; s > 0; s >>= 1) {
        if (tid < s) { s1[tid] += s1[tid + s]; s2[tid] += s2[tid + s]; }
        __syncthreads();
    }
    if (tid == 0) {
        const float mean = s1[0] / (float)Mtok;
        const float var  = s2[0] / (float)Mtok - mean * mean;
        const float sc   = gamma[c] * rsqrtf(var + 1e-5f);
        st[c] = sc; st[Hsz + c] = beta[c] - mean * sc;
    }
}

__global__ void bn_apply(const __bf16* __restrict__ in, __bf16* __restrict__ o,
                         const float* __restrict__ st, size_t n) {
    for (size_t i = (size_t)blockIdx.x * blockDim.x + threadIdx.x; i < n;
         i += (size_t)gridDim.x * blockDim.x) {
        const int c = (int)(i & (Hsz - 1));
        o[i] = f2bf(bf2f(in[i]) * st[c] + st[Hsz + c]);
    }
}

__global__ void f32_to_bf16(const float* __restrict__ s, __bf16* __restrict__ d,
                            size_t n, float scale) {
    for (size_t i = (size_t)blockIdx.x * blockDim.x + threadIdx.x; i < n;
         i += (size_t)gridDim.x * blockDim.x)
        d[i] = f2bf(s[i] * scale);
}

__global__ void scale_copy_f32(const float* __restrict__ s, float* __restrict__ d,
                               int n, float scale) {
    const int i = blockIdx.x * blockDim.x + threadIdx.x;
    if (i < n) d[i] = s[i] * scale;
}

__global__ void fill_kernel(float* __restrict__ p, size_t n, float v) {
    for (size_t i = (size_t)blockIdx.x * blockDim.x + threadIdx.x; i < n;
         i += (size_t)gridDim.x * blockDim.x)
        p[i] = v;
}

// ---------------------------------------------------------------------------
extern "C" void kernel_launch(void* const* d_in, const int* in_sizes, int n_in,
                              void* d_out, int out_size, void* d_ws, size_t ws_size,
                              hipStream_t stream) {
    (void)in_sizes; (void)n_in; (void)out_size; (void)ws_size;
    const float* x = (const float*)d_in[0];
    const float* e0f[4] = {(const float*)d_in[1],  (const float*)d_in[2],  (const float*)d_in[3],  (const float*)d_in[4]};
    const float* e0b[4] = {(const float*)d_in[5],  (const float*)d_in[6],  (const float*)d_in[7],  (const float*)d_in[8]};
    const float* e1f[4] = {(const float*)d_in[9],  (const float*)d_in[10], (const float*)d_in[11], (const float*)d_in[12]};
    const float* e1b[4] = {(const float*)d_in[13], (const float*)d_in[14], (const float*)d_in[15], (const float*)d_in[16]};
    const float* w_in  = (const float*)d_in[17];
    const float* b_in  = (const float*)d_in[18];
    const float* w_out = (const float*)d_in[19];
    const float* b_out = (const float*)d_in[20];
    const float* dc0[4] = {(const float*)d_in[21], (const float*)d_in[22], (const float*)d_in[23], (const float*)d_in[24]};
    const float* dc1[4] = {(const float*)d_in[25], (const float*)d_in[26], (const float*)d_in[27], (const float*)d_in[28]};
    const float* gamma = (const float*)d_in[29];
    const float* beta  = (const float*)d_in[30];
    const float* out_w = (const float*)d_in[31];
    const float* out_b = (const float*)d_in[32];
    const float* se_w[3] = {(const float*)d_in[33], (const float*)d_in[35], (const float*)d_in[37]};
    const float* se_b[3] = {(const float*)d_in[34], (const float*)d_in[36], (const float*)d_in[38]};
    const float* sd_w[3] = {(const float*)d_in[39], (const float*)d_in[41], (const float*)d_in[43]};
    const float* sd_b[3] = {(const float*)d_in[40], (const float*)d_in[42], (const float*)d_in[44]};
    const float* fu_w = (const float*)d_in[45];
    const float* fu_b = (const float*)d_in[46];

    // ---- workspace layout ----
    __bf16* WB = (__bf16*)d_ws;           // bf16 weight arena (~1.31M elems)
    size_t off = 1310720;                 // arena size (all weights, padded)
    __bf16* xb = WB + off; off += (size_t)Mtok * 64;
    __bf16* G  = WB + off; off += (size_t)Mtok * 512;
    __bf16* Q  = WB + off; off += (size_t)Mtok * 768;
    __bf16* X1 = WB + off; off += (size_t)Mtok * 256;
    __bf16* X2 = WB + off; off += (size_t)Mtok * 256;
    __bf16* XF = WB + off; off += (size_t)Mtok * 128;
    float*  ST = (float*)(WB + off);      // 256 floats BN scale/shift
    float*  BQ = ST + 256;                // 768 floats scaled b_in

    float* hybrid = (float*)d_out;                 // [B,S,F]
    float* comb   = hybrid + (size_t)Mtok * Fsz;   // [B,S,S]

    const size_t SCAN_SMEM = 131072 + 4096 + 32768 + 8192;  // 176128 B
    const size_t ATT_SMEM  = 131072;

    // weight arena sub-allocation (element counts all multiples of 8)
    size_t wo = 0;
    auto wslot = [&](size_t n) { __bf16* p = WB + wo; wo += n; return p; };
    __bf16* Wb_e0f_ih = wslot(512 * 64);   __bf16* Wb_e0f_hh = wslot(512 * 128);
    __bf16* Wb_e0b_ih = wslot(512 * 64);   __bf16* Wb_e0b_hh = wslot(512 * 128);
    __bf16* Wb_e1f_ih = wslot(512 * 256);  __bf16* Wb_e1f_hh = wslot(512 * 128);
    __bf16* Wb_e1b_ih = wslot(512 * 256);  __bf16* Wb_e1b_hh = wslot(512 * 128);
    __bf16* Wb_win    = wslot(768 * 256);  __bf16* Wb_wout   = wslot(256 * 256);
    __bf16* Wb_d0_ih  = wslot(512 * 256);  __bf16* Wb_d0_hh  = wslot(512 * 128);
    __bf16* Wb_d1_ih  = wslot(512 * 128);  __bf16* Wb_d1_hh  = wslot(512 * 128);
    __bf16* Wb_out    = wslot(64 * 128);
    __bf16* Wb_se0 = wslot(256 * 64); __bf16* Wb_se1 = wslot(128 * 256); __bf16* Wb_se2 = wslot(64 * 128);
    __bf16* Wb_sd0 = wslot(128 * 64); __bf16* Wb_sd1 = wslot(256 * 128); __bf16* Wb_sd2 = wslot(64 * 256);
    __bf16* Wb_fu  = wslot(64 * 128);

    auto cvt = [&](const float* s, __bf16* d, size_t n, float sc = 1.0f) {
        const int blocks = (int)((n + 256 * 8 - 1) / (256 * 8));
        f32_to_bf16<<<blocks, 256, 0, stream>>>(s, d, n, sc);
    };
    // ---- weight + input conversions (q projection pre-scaled by 1/sqrt(dh)) ----
    cvt(x, xb, (size_t)Mtok * 64);
    cvt(e0f[0], Wb_e0f_ih, 512 * 64);  cvt(e0f[1], Wb_e0f_hh, 512 * 128);
    cvt(e0b[0], Wb_e0b_ih, 512 * 64);  cvt(e0b[1], Wb_e0b_hh, 512 * 128);
    cvt(e1f[0], Wb_e1f_ih, 512 * 256); cvt(e1f[1], Wb_e1f_hh, 512 * 128);
    cvt(e1b[0], Wb_e1b_ih, 512 * 256); cvt(e1b[1], Wb_e1b_hh, 512 * 128);
    cvt(w_in, Wb_win, 256 * 256, 0.125f);                       // q rows scaled
    cvt(w_in + 256 * 256, Wb_win + 256 * 256, 512 * 256, 1.0f); // k,v rows
    cvt(w_out, Wb_wout, 256 * 256);
    cvt(dc0[0], Wb_d0_ih, 512 * 256);  cvt(dc0[1], Wb_d0_hh, 512 * 128);
    cvt(dc1[0], Wb_d1_ih, 512 * 128);  cvt(dc1[1], Wb_d1_hh, 512 * 128);
    cvt(out_w, Wb_out, 64 * 128);
    cvt(se_w[0], Wb_se0, 256 * 64); cvt(se_w[1], Wb_se1, 128 * 256); cvt(se_w[2], Wb_se2, 64 * 128);
    cvt(sd_w[0], Wb_sd0, 128 * 64); cvt(sd_w[1], Wb_sd1, 256 * 128); cvt(sd_w[2], Wb_sd2, 64 * 256);
    cvt(fu_w, Wb_fu, 64 * 128);
    scale_copy_f32<<<1, 256, 0, stream>>>(b_in, BQ, 256, 0.125f);
    scale_copy_f32<<<2, 256, 0, stream>>>(b_in + 256, BQ + 256, 512, 1.0f);

    auto gemm = [&](const __bf16* A, int lda, const __bf16* W,
                    const float* b1, const float* b2,
                    void* C, int ldc, int coff, int N_, int K_, int relu, int obf) {
        dim3 grid(N_ / 64, Mtok / 64);
        gemm_wmma<<<grid, 128, 0, stream>>>(A, lda, W, b1, b2, C, ldc, coff, K_, relu, obf);
    };
    auto scan = [&](const __bf16* g, const __bf16* whh, __bf16* o, int ldo, int coff, int rev) {
        lstm_scan<<<Bsz / 16, 256, SCAN_SMEM, stream>>>(g, whh, o, ldo, coff, rev);
    };

    // ---- bidirectional LSTM encoder ----
    gemm(xb, Fsz, Wb_e0f_ih, e0f[2], e0f[3], G, 512, 0, 512, Fsz, 0, 1);
    scan(G, Wb_e0f_hh, X1, Esz, 0, 0);
    gemm(xb, Fsz, Wb_e0b_ih, e0b[2], e0b[3], G, 512, 0, 512, Fsz, 0, 1);
    scan(G, Wb_e0b_hh, X1, Esz, Hsz, 1);
    gemm(X1, Esz, Wb_e1f_ih, e1f[2], e1f[3], G, 512, 0, 512, Esz, 0, 1);
    scan(G, Wb_e1f_hh, X2, Esz, 0, 0);
    gemm(X1, Esz, Wb_e1b_ih, e1b[2], e1b[3], G, 512, 0, 512, Esz, 0, 1);
    scan(G, Wb_e1b_hh, X2, Esz, Hsz, 1);

    // ---- attention ----
    gemm(X2, Esz, Wb_win, BQ, nullptr, Q, 3 * Esz, 0, 3 * Esz, Esz, 0, 1);
    fill_kernel<<<4096, 256, 0, stream>>>(comb, (size_t)Mtok * Ssz, 0.4f / (float)Ssz);
    attn_fused<<<dim3(2, NHsz, Bsz), 256, ATT_SMEM, stream>>>(Q, X1, comb);
    gemm(X1, Esz, Wb_wout, b_out, nullptr, X2, Esz, 0, Esz, Esz, 0, 1);

    // ---- unidirectional LSTM decoder ----
    gemm(X2, Esz, Wb_d0_ih, dc0[2], dc0[3], G, 512, 0, 512, Esz, 0, 1);
    scan(G, Wb_d0_hh, X1, Hsz, 0, 0);
    gemm(X1, Hsz, Wb_d1_ih, dc1[2], dc1[3], G, 512, 0, 512, Hsz, 0, 1);
    scan(G, Wb_d1_hh, X2, Hsz, 0, 0);

    // ---- BatchNorm (stats + folded apply) then output projection -> XF[:,0:64) ----
    bn_stats<<<Hsz, 256, 0, stream>>>(X2, gamma, beta, ST);
    bn_apply<<<4096, 256, 0, stream>>>(X2, X1, ST, (size_t)Mtok * Hsz);
    gemm(X1, Hsz, Wb_out, out_b, nullptr, XF, 2 * Fsz, 0, Fsz, Hsz, 0, 1);

    // ---- single-log MLP path -> XF[:,64:128) ----
    gemm(xb, Fsz, Wb_se0, se_b[0], nullptr, X1, Esz, 0, Esz, Fsz, 1, 1);
    gemm(X1, Esz, Wb_se1, se_b[1], nullptr, X2, Hsz, 0, Hsz, Esz, 1, 1);
    gemm(X2, Hsz, Wb_se2, se_b[2], nullptr, X1, Fsz, 0, Fsz, Hsz, 0, 1);
    gemm(X1, Fsz, Wb_sd0, sd_b[0], nullptr, X2, Hsz, 0, Hsz, Fsz, 1, 1);
    gemm(X2, Hsz, Wb_sd1, sd_b[1], nullptr, X1, Esz, 0, Esz, Hsz, 1, 1);
    gemm(X1, Esz, Wb_sd2, sd_b[2], nullptr, XF, 2 * Fsz, Fsz, Fsz, Esz, 0, 1);

    // ---- fusion -> hybrid output (fp32) ----
    gemm(XF, 2 * Fsz, Wb_fu, fu_b, nullptr, hybrid, Fsz, 0, Fsz, 2 * Fsz, 0, 0);
}
